// Attention_80831284511385
// MI455X (gfx1250) — compile-verified, matrix operand order checked by reading, not run on hardware
//
#include <hip/hip_runtime.h>
#include <hip/hip_bf16.h>

// ---------------------------------------------------------------------------
// GQA attention (B=1, S=2048, DIM=4096, NH=32, NKV=8, HD=128) on gfx1250.
// Matmuls: v_wmma_f32_16x16x32_bf16 (wave32). Flash attention stages KV tiles
// through LDS with global_load_async_to_lds_b128 (ASYNCcnt), double-buffered.
// ---------------------------------------------------------------------------

#define DIM 4096
#define NH  32
#define NKV 8
#define HD  128
#define SEQ 2048
#define NREP (NH / NKV)

typedef __bf16  v16bf __attribute__((ext_vector_type(16)));
typedef __bf16  bf16x8 __attribute__((ext_vector_type(8)));
typedef float   v8f   __attribute__((ext_vector_type(8)));

static __device__ __forceinline__ v8f wmma_bf16(v16bf a, v16bf b, v8f c) {
    // v_wmma_f32_16x16x32_bf16 : D = A(16x32) * B(32x16) + C(16x16 f32)
    return __builtin_amdgcn_wmma_f32_16x16x32_bf16(
        /*neg_a=*/false, a, /*neg_b=*/false, b,
        /*c_mod=*/(short)0, c, /*reuse_a=*/false, /*reuse_b=*/false);
}

// Load a 16x32 bf16 "A-style" fragment from a row-major matrix (global or LDS).
// Lane l holds row (row0 + l%16); half = l/16 selects the K sub-blocks:
//   VGPR0..3 : K = half*8 + 0..7      (contiguous 16 bytes)
//   VGPR4..7 : K = 16 + half*8 + 0..7 (contiguous 16 bytes)
static __device__ __forceinline__ v16bf load_frag(const __bf16* p, int row0, int k0,
                                                  int ld, int lane) {
    const int r = lane & 15, h = lane >> 4;
    const __bf16* q = p + (size_t)(row0 + r) * (size_t)ld + k0 + h * 8;
    union { v16bf v; bf16x8 h2[2]; } u;
    u.h2[0] = *(const bf16x8*)(q);
    u.h2[1] = *(const bf16x8*)(q + 16);
    return u.v;
}

// Async copy of 16 bytes/lane from global to LDS (tracked by ASYNCcnt).
static __device__ __forceinline__ void async_cp16(unsigned lds_off, const __bf16* g) {
    asm volatile("global_load_async_to_lds_b128 %0, %1, off"
                 :: "v"(lds_off), "v"(g) : "memory");
}

// ---------------------------------------------------------------------------
// Elementwise helpers
// ---------------------------------------------------------------------------
__global__ void cvt_f32_to_bf16(const float* __restrict__ src,
                                __bf16* __restrict__ dst, int n) {
    int i = blockIdx.x * blockDim.x + threadIdx.x;
    if (i < n) dst[i] = (__bf16)src[i];
}

// src is (K rows x N cols) fp32 row-major; dst is (N x K) bf16 row-major.
__global__ void transpose_to_bf16(const float* __restrict__ src,
                                  __bf16* __restrict__ dst, int K, int N) {
    int i = blockIdx.x * blockDim.x + threadIdx.x;
    if (i >= K * N) return;
    int k = i / N, n = i - k * N;
    dst[(size_t)n * K + k] = (__bf16)src[i];
}

// RoPE: src is (S x (H*128)) fp32; dst is (H x S x 128) bf16.
// Pair (i, i+64) rotated by angle s * 10000^(-i/64).
__global__ void rope_to_heads(const float* __restrict__ src,
                              __bf16* __restrict__ dst, int H) {
    int t = blockIdx.x * blockDim.x + threadIdx.x;
    int total = SEQ * H * 64;
    if (t >= total) return;
    int i = t & 63;
    int h = (t >> 6) % H;
    int s = t / (64 * H);
    float inv_freq = __powf(10000.0f, -(float)i * (1.0f / 64.0f));
    float ang = (float)s * inv_freq;
    float sn, cs;
    __sincosf(ang, &sn, &cs);
    const float* row = src + (size_t)s * (H * HD) + h * HD;
    float x1 = row[i], x2 = row[i + 64];
    __bf16* drow = dst + ((size_t)h * SEQ + s) * HD;
    drow[i]      = (__bf16)(x1 * cs - x2 * sn);
    drow[i + 64] = (__bf16)(x2 * cs + x1 * sn);
}

// V transpose: src (S x (NKV*128)) fp32 -> dst (NKV x 128 x S) bf16.
__global__ void v_transpose(const float* __restrict__ src,
                            __bf16* __restrict__ dst) {
    int t = blockIdx.x * blockDim.x + threadIdx.x;
    int total = SEQ * NKV * HD;
    if (t >= total) return;
    int s = t & (SEQ - 1);
    int d = (t / SEQ) & (HD - 1);
    int g = t / (SEQ * HD);
    dst[((size_t)g * HD + d) * SEQ + s] = (__bf16)src[(size_t)s * (NKV * HD) + g * HD + d];
}

// ---------------------------------------------------------------------------
// WMMA GEMM: C(MxN,f32) = A(MxK,bf16 row-major) * BT(NxK,bf16 row-major)^T
// Block = 128 threads (4 waves), block tile 64x128, wave tile 32x64.
// Per K=32 step: 6 fragment loads (192 B/lane) feed 8 WMMAs -> 21.3 FLOP/B.
// ---------------------------------------------------------------------------
__global__ __launch_bounds__(128) void gemm_bf16_wmma(
    const __bf16* __restrict__ A, const __bf16* __restrict__ BT,
    float* __restrict__ C, int M, int N, int K) {
    const int lane = threadIdx.x & 31;
    const int w    = threadIdx.x >> 5;
    const int m0 = blockIdx.y * 64  + (w >> 1) * 32;
    const int n0 = blockIdx.x * 128 + (w & 1) * 64;

    v8f acc0[4], acc1[4];
#pragma unroll
    for (int j = 0; j < 4; ++j) {
        acc0[j] = (v8f){0,0,0,0,0,0,0,0};
        acc1[j] = (v8f){0,0,0,0,0,0,0,0};
    }

    for (int k0 = 0; k0 < K; k0 += 32) {
        // branch-free speculative prefetch one super-step ahead
        __builtin_prefetch(A  + (size_t)(m0 + (lane & 15)) * K + k0 + 128, 0, 3);
        __builtin_prefetch(BT + (size_t)(n0 + (lane & 15)) * K + k0 + 128, 0, 3);
        v16bf a0 = load_frag(A, m0,      k0, K, lane);
        v16bf a1 = load_frag(A, m0 + 16, k0, K, lane);
        v16bf b[4];
#pragma unroll
        for (int j = 0; j < 4; ++j) b[j] = load_frag(BT, n0 + j * 16, k0, K, lane);
#pragma unroll
        for (int j = 0; j < 4; ++j) {
            acc0[j] = wmma_bf16(a0, b[j], acc0[j]);
            acc1[j] = wmma_bf16(a1, b[j], acc1[j]);
        }
    }

    // C-layout: VGPR v -> row (v + 8*half), column = lane&15 within 16-wide tile
    const int half = lane >> 4, cl = lane & 15;
#pragma unroll
    for (int v = 0; v < 8; ++v) {
        int r0 = m0 + v + 8 * half;
#pragma unroll
        for (int j = 0; j < 4; ++j) {
            C[(size_t)r0 * N + (n0 + j * 16 + cl)]        = acc0[j][v];
            C[(size_t)(r0 + 16) * N + (n0 + j * 16 + cl)] = acc1[j][v];
        }
    }
}

// ---------------------------------------------------------------------------
// Flash attention (causal, GQA). One block = (64 q rows, 1 head). 4 waves,
// each wave owns 16 q rows. KV processed in 32-column tiles which are staged
// into LDS once per block via async-to-LDS (double-buffered, 1 tile ahead).
// Q: (NH x S x 128) bf16, K: (NKV x S x 128) bf16, Vt: (NKV x 128 x S) bf16.
// Output: (S x NH*128) bf16.
// ---------------------------------------------------------------------------
static __device__ __forceinline__ float half_max(float v) {
#pragma unroll
    for (int m = 1; m < 16; m <<= 1) v = fmaxf(v, __shfl_xor(v, m, 32));
    return v;
}
static __device__ __forceinline__ float half_sum(float v) {
#pragma unroll
    for (int m = 1; m < 16; m <<= 1) v += __shfl_xor(v, m, 32);
    return v;
}

__global__ __launch_bounds__(128) void flash_attn_wmma(
    const __bf16* __restrict__ Q, const __bf16* __restrict__ Kh,
    const __bf16* __restrict__ Vt, __bf16* __restrict__ Oout) {
    // Double-buffered KV tiles + per-wave P staging. 36 KB total.
    __shared__ alignas(16) __bf16 Ktile[2][32 * 128];   // rows c0..c0+31, 8 KB each
    __shared__ alignas(16) __bf16 Vtile[2][128 * 32];   // [d][kv-col], 8 KB each
    __shared__ alignas(16) __bf16 pstage[4][16 * 32];

    const int lane = threadIdx.x & 31;
    const int w    = threadIdx.x >> 5;
    const int h    = blockIdx.y;
    const int g    = h >> 2;            // h / NREP
    const int qr0  = blockIdx.x * 64 + w * 16;
    const int half = lane >> 4, cl = lane & 15;

    const __bf16* Qh = Q  + (size_t)h * SEQ * HD;
    const __bf16* Kg = Kh + (size_t)g * SEQ * HD;
    const __bf16* Vg = Vt + (size_t)g * HD * SEQ;

    v16bf qf[4];
#pragma unroll
    for (int d = 0; d < 4; ++d) qf[d] = load_frag(Qh, qr0, d * 32, HD, lane);

    v8f o[8];
    float mrow[8], lrow[8];
#pragma unroll
    for (int b = 0; b < 8; ++b) o[b] = (v8f){0,0,0,0,0,0,0,0};
#pragma unroll
    for (int v = 0; v < 8; ++v) { mrow[v] = -1e30f; lrow[v] = 0.0f; }

    const float scale = 0.08838834764831845f;  // 1/sqrt(128)
    const int ntiles = blockIdx.x * 2 + 2;     // covers cols < qbase+64 (block-uniform)

    // Issue the async tile copy for kv tile jt into buffer buf.
    // 8 instructions per wave: 4 for K (contiguous 16 KB? no: 8 KB), 4 for V.
    auto issue_tile = [&](int jt, int buf) {
        const int c0 = jt * 32;
        const unsigned kbase = (unsigned)(uintptr_t)(&Ktile[buf][0]);
        const unsigned vbase = (unsigned)(uintptr_t)(&Vtile[buf][0]);
#pragma unroll
        for (int i = 0; i < 4; ++i) {
            // K tile: rows c0..c0+31 x 128 cols, contiguous 8 KB region.
            int ck = i * 128 + w * 32 + lane;             // 16B chunk id 0..511
            async_cp16(kbase + ck * 16, Kg + (size_t)c0 * HD + ck * 8);
            // V tile: rows d=0..127, cols c0..c0+31 (strided in global).
            int cv = i * 128 + w * 32 + lane;             // chunk id 0..511
            int vrow = cv >> 2, vcc = cv & 3;
            async_cp16(vbase + cv * 16, Vg + (size_t)vrow * SEQ + c0 + vcc * 8);
        }
    };

    issue_tile(0, 0);

    for (int jt = 0; jt < ntiles; ++jt) {
        const int c0 = jt * 32;
        const int buf = jt & 1;
        const bool have_next = (jt + 1 < ntiles);
        if (have_next) {
            issue_tile(jt + 1, buf ^ 1);
            // 16 outstanding; wait until only the 8 just-issued remain.
            asm volatile("s_wait_asynccnt 0x8" ::: "memory");
        } else {
            asm volatile("s_wait_asynccnt 0x0" ::: "memory");
        }
        __syncthreads();   // all waves' copies of tile jt are in LDS

        // S = Q * K^T  (two 16-wide column tiles) from LDS
        v8f s0 = (v8f){0,0,0,0,0,0,0,0}, s1 = s0;
#pragma unroll
        for (int d = 0; d < 4; ++d) {
            v16bf k0f = load_frag(&Ktile[buf][0], 0,  d * 32, HD, lane);
            v16bf k1f = load_frag(&Ktile[buf][0], 16, d * 32, HD, lane);
            s0 = wmma_bf16(qf[d], k0f, s0);
            s1 = wmma_bf16(qf[d], k1f, s1);
        }
        // Online softmax per row, causal mask in registers.
#pragma unroll
        for (int v = 0; v < 8; ++v) {
            const int row = qr0 + v + 8 * half;
            float a0 = s0[v] * scale; if (c0 + cl      > row) a0 = -1e30f;
            float a1 = s1[v] * scale; if (c0 + 16 + cl > row) a1 = -1e30f;
            float mnew = fmaxf(mrow[v], half_max(fmaxf(a0, a1)));
            float p0 = __expf(a0 - mnew);
            float p1 = __expf(a1 - mnew);
            float f  = __expf(mrow[v] - mnew);
            mrow[v] = mnew;
            lrow[v] = lrow[v] * f + half_sum(p0 + p1);
#pragma unroll
            for (int b = 0; b < 8; ++b) o[b][v] *= f;
            pstage[w][(v + 8 * half) * 32 + cl]      = (__bf16)p0;
            pstage[w][(v + 8 * half) * 32 + 16 + cl] = (__bf16)p1;
        }
        // Stage P through LDS to get an A-layout fragment (per-wave region;
        // same-wave DS ops are in-order, wait for visibility anyway).
        asm volatile("s_wait_dscnt 0x0" ::: "memory");
        __builtin_amdgcn_wave_barrier();
        union { v16bf v; bf16x8 h2[2]; } up;
        const __bf16* pp = &pstage[w][cl * 32 + half * 8];
        up.h2[0] = *(const bf16x8*)(pp);
        up.h2[1] = *(const bf16x8*)(pp + 16);
        v16bf pf = up.v;
        // O += P * V   (V^T rows are the d-dimension) from LDS
#pragma unroll
        for (int b = 0; b < 8; ++b) {
            v16bf vf = load_frag(&Vtile[buf][0], b * 16, 0, 32, lane);
            o[b] = wmma_bf16(pf, vf, o[b]);
        }
        __syncthreads();   // all reads of buffer `buf` done before it is refilled
    }

    // Normalize and store bf16 output (S x NH*128)
#pragma unroll
    for (int v = 0; v < 8; ++v) {
        const float inv = 1.0f / lrow[v];
        const int srow = qr0 + v + 8 * half;
        __bf16* orow = Oout + (size_t)srow * (NH * HD) + h * HD;
#pragma unroll
        for (int b = 0; b < 8; ++b)
            orow[b * 16 + cl] = (__bf16)(o[b][v] * inv);
    }
}

// ---------------------------------------------------------------------------
// Host-side launch
// ---------------------------------------------------------------------------
extern "C" void kernel_launch(void* const* d_in, const int* in_sizes, int n_in,
                              void* d_out, int out_size, void* d_ws, size_t ws_size,
                              hipStream_t stream) {
    const float* x  = (const float*)d_in[0];   // (S, DIM)
    const float* wq = (const float*)d_in[1];   // (DIM, 4096)
    const float* wk = (const float*)d_in[2];   // (DIM, 1024)
    const float* wv = (const float*)d_in[3];   // (DIM, 1024)
    const float* wo = (const float*)d_in[4];   // (4096, DIM)
    float* out = (float*)d_out;                // (S, DIM)

    char* ws = (char*)d_ws;
    size_t off = 0;
    auto take = [&](size_t bytes) { size_t o = off; off = (off + bytes + 255) & ~(size_t)255; return o; };

    __bf16* xbf  = (__bf16*)(ws + take((size_t)SEQ * DIM * 2));          // 16 MB
    __bf16* wqT  = (__bf16*)(ws + take((size_t)DIM * 4096 * 2));         // 32 MB
    __bf16* wkT  = (__bf16*)(ws + take((size_t)DIM * 1024 * 2));         //  8 MB
    __bf16* wvT  = (__bf16*)(ws + take((size_t)DIM * 1024 * 2));         //  8 MB
    __bf16* woT  = (__bf16*)(ws + take((size_t)4096 * DIM * 2));         // 32 MB
    float*  qf   = (float*) (ws + take((size_t)SEQ * 4096 * 4));         // 32 MB
    float*  kf   = (float*) (ws + take((size_t)SEQ * 1024 * 4));         //  8 MB
    float*  vf   = (float*) (ws + take((size_t)SEQ * 1024 * 4));         //  8 MB
    __bf16* Qbf  = (__bf16*)(ws + take((size_t)NH  * SEQ * HD * 2));     // 16 MB
    __bf16* Kbf  = (__bf16*)(ws + take((size_t)NKV * SEQ * HD * 2));     //  4 MB
    __bf16* Vtb  = (__bf16*)(ws + take((size_t)NKV * HD * SEQ * 2));     //  4 MB
    __bf16* attn = (__bf16*)qf;  // alias: qf dead after RoPE, reuse for attn out
    (void)ws_size; (void)n_in; (void)in_sizes; (void)out_size;

    const int T = 256;
    // 1) precision conversion + weight transposes
    cvt_f32_to_bf16<<<(SEQ * DIM + T - 1) / T, T, 0, stream>>>(x, xbf, SEQ * DIM);
    transpose_to_bf16<<<(DIM * 4096 + T - 1) / T, T, 0, stream>>>(wq, wqT, DIM, 4096);
    transpose_to_bf16<<<(DIM * 1024 + T - 1) / T, T, 0, stream>>>(wk, wkT, DIM, 1024);
    transpose_to_bf16<<<(DIM * 1024 + T - 1) / T, T, 0, stream>>>(wv, wvT, DIM, 1024);
    transpose_to_bf16<<<(4096 * DIM + T - 1) / T, T, 0, stream>>>(wo, woT, 4096, DIM);

    // 2) QKV projections (WMMA)
    gemm_bf16_wmma<<<dim3(4096 / 128, SEQ / 64), 128, 0, stream>>>(xbf, wqT, qf, SEQ, 4096, DIM);
    gemm_bf16_wmma<<<dim3(1024 / 128, SEQ / 64), 128, 0, stream>>>(xbf, wkT, kf, SEQ, 1024, DIM);
    gemm_bf16_wmma<<<dim3(1024 / 128, SEQ / 64), 128, 0, stream>>>(xbf, wvT, vf, SEQ, 1024, DIM);

    // 3) RoPE + layout for attention
    rope_to_heads<<<(SEQ * NH * 64 + T - 1) / T, T, 0, stream>>>(qf, Qbf, NH);
    rope_to_heads<<<(SEQ * NKV * 64 + T - 1) / T, T, 0, stream>>>(kf, Kbf, NKV);
    v_transpose<<<(SEQ * NKV * HD + T - 1) / T, T, 0, stream>>>(vf, Vtb);

    // 4) causal flash attention (WMMA + async-to-LDS KV staging)
    flash_attn_wmma<<<dim3(SEQ / 64, NH), 128, 0, stream>>>(Qbf, Kbf, Vtb, attn);

    // 5) output projection (WMMA) -> f32 result
    gemm_bf16_wmma<<<dim3(DIM / 128, SEQ / 64), 128, 0, stream>>>(attn, woT, out, SEQ, DIM, 4096);
}
